// LSTMModel_38276748542217
// MI455X (gfx1250) — compile-verified, matrix operand order checked by reading, not run on hardware
//
#include <hip/hip_runtime.h>

typedef __attribute__((ext_vector_type(16))) _Float16 v16h;
typedef __attribute__((ext_vector_type(8)))  _Float16 v8h;
typedef __attribute__((ext_vector_type(8)))  float    v8f;

// Problem sizes (fixed by the reference)
constexpr int Bsz  = 4096;
constexpr int Tlen = 256;
constexpr int Fin  = 64;

// LDS layout (f16 weights with padded strides for bank spread; strides kept
// multiples of 8 halves so v8h (16B) loads stay aligned)
constexpr int S_W1  = 72;            // stride for 64-wide weight rows (halves)
constexpr int S_W2s = 40;            // stride for 32-wide weight rows (W_hh2)
constexpr int S_ST1 = 80;            // stride for 16x64 h1 staging (halves)
constexpr int S_ST2 = 40;            // stride for 16x32 h2 staging (halves)
constexpr int S_XB  = 72;            // stride for x tile rows (floats) = 288B

constexpr int N_WIH1 = 256 * S_W1;
constexpr int N_WHH1 = 256 * S_W1;
constexpr int N_WIH2 = 128 * S_W1;
constexpr int N_WHH2 = 128 * S_W2s;
constexpr int WAVES  = 4;
constexpr int N_ST1  = WAVES * 16 * S_ST1;
constexpr int N_ST2  = WAVES * 16 * S_ST2;
constexpr int N_FIN  = WAVES * 16 * 32;            // floats
constexpr int N_XB   = WAVES * 2 * 16 * S_XB;      // floats (double-buffered)
constexpr size_t SMEM_BYTES =
    (size_t)(N_WIH1 + N_WHH1 + N_WIH2 + N_WHH2 + N_ST1 + N_ST2) * sizeof(_Float16)
  + (size_t)(N_FIN + N_XB) * sizeof(float);

// Byte strides of x in global memory
constexpr int XROW_BYTES = Tlen * Fin * 4;   // 65536
constexpr int XT_BYTES   = Fin * 4;          // 256

__device__ __forceinline__ float sigf(float x) {
  return 1.0f / (1.0f + __expf(-x));
}
__device__ __forceinline__ float tanh_fast(float x) {
  x = fminf(fmaxf(x, -15.0f), 15.0f);
  float e = __expf(2.0f * x);
  return (e - 1.0f) / (e + 1.0f);
}

union FragU {
  v16h v;
  v8h  h8[2];
  _Float16 e[16];
};

// A/B fragment (16-bit, 16x16x32 shape) from an LDS f16 array with row stride
// `rowlen` halves. CDNA5 layout: elements 0..7 hold K = half*8..+7,
// elements 8..15 hold K = 16 + half*8..+7 (two 16B runs -> ds_load_b128).
__device__ __forceinline__ v16h frag_from_lds(const _Float16* base, int rowlen,
                                              int row, int k0, int half) {
  FragU u;
  const _Float16* p = base + row * rowlen + k0 + half * 8;
  u.h8[0] = *(const v8h*)(p);
  u.h8[1] = *(const v8h*)(p + 16);
  return u.v;
}

__device__ __forceinline__ v16h pack16(float4 a, float4 b, float4 c, float4 d) {
  FragU u;
  u.e[0]  = (_Float16)a.x; u.e[1]  = (_Float16)a.y;
  u.e[2]  = (_Float16)a.z; u.e[3]  = (_Float16)a.w;
  u.e[4]  = (_Float16)b.x; u.e[5]  = (_Float16)b.y;
  u.e[6]  = (_Float16)b.z; u.e[7]  = (_Float16)b.w;
  u.e[8]  = (_Float16)c.x; u.e[9]  = (_Float16)c.y;
  u.e[10] = (_Float16)c.z; u.e[11] = (_Float16)c.w;
  u.e[12] = (_Float16)d.x; u.e[13] = (_Float16)d.y;
  u.e[14] = (_Float16)d.z; u.e[15] = (_Float16)d.w;
  return u.v;
}

// x A-fragment from the LDS f32 staging tile (filled by async loads).
__device__ __forceinline__ v16h frag_x_from_lds(const float* base, int row,
                                                int k0, int half) {
  const float* p = base + row * S_XB + k0 + half * 8;
  const float4* r0 = (const float4*)(p);
  const float4* r1 = (const float4*)(p + 16);
  return pack16(r0[0], r0[1], r1[0], r1[1]);
}

__device__ __forceinline__ v8f wmma_f16(v16h a, v16h b, v8f c) {
  return __builtin_amdgcn_wmma_f32_16x16x32_f16(
      false, a, false, b, (short)0, c, false, false);
}

// Issue one async 16-row x-tile fill: 8 x global_load_async_to_lds_b128.
// Per the ISA, INST_OFFSET advances both the LDS and global byte addresses,
// so a single per-lane LDS addr + per-lane global offset covers the tile.
__device__ __forceinline__ void async_x_tile(const void* gbase /*wave-uniform*/,
                                             unsigned ldsaddr /*per-lane*/,
                                             unsigned goff /*per-lane*/) {
  asm volatile("global_load_async_to_lds_b128 %0, %1, %2 offset:0"
               :: "v"(ldsaddr), "v"(goff), "s"(gbase) : "memory");
  asm volatile("global_load_async_to_lds_b128 %0, %1, %2 offset:32"
               :: "v"(ldsaddr), "v"(goff), "s"(gbase) : "memory");
  asm volatile("global_load_async_to_lds_b128 %0, %1, %2 offset:64"
               :: "v"(ldsaddr), "v"(goff), "s"(gbase) : "memory");
  asm volatile("global_load_async_to_lds_b128 %0, %1, %2 offset:96"
               :: "v"(ldsaddr), "v"(goff), "s"(gbase) : "memory");
  asm volatile("global_load_async_to_lds_b128 %0, %1, %2 offset:128"
               :: "v"(ldsaddr), "v"(goff), "s"(gbase) : "memory");
  asm volatile("global_load_async_to_lds_b128 %0, %1, %2 offset:160"
               :: "v"(ldsaddr), "v"(goff), "s"(gbase) : "memory");
  asm volatile("global_load_async_to_lds_b128 %0, %1, %2 offset:192"
               :: "v"(ldsaddr), "v"(goff), "s"(gbase) : "memory");
  asm volatile("global_load_async_to_lds_b128 %0, %1, %2 offset:224"
               :: "v"(ldsaddr), "v"(goff), "s"(gbase) : "memory");
}

__device__ __forceinline__ void wait_async0() {
  asm volatile("s_wait_asynccnt 0" ::: "memory");
}

__global__ __launch_bounds__(128)
void lstm2_fused_wmma(const float* __restrict__ x,
                      const float* __restrict__ W_ih1, const float* __restrict__ W_hh1,
                      const float* __restrict__ b_ih1, const float* __restrict__ b_hh1,
                      const float* __restrict__ W_ih2, const float* __restrict__ W_hh2,
                      const float* __restrict__ b_ih2, const float* __restrict__ b_hh2,
                      const float* __restrict__ W_fc1, const float* __restrict__ b_fc1,
                      const float* __restrict__ W_fc2, const float* __restrict__ b_fc2,
                      const float* __restrict__ W_out, const float* __restrict__ b_out,
                      float* __restrict__ out) {
  extern __shared__ char smem[];
  _Float16* sWih1 = (_Float16*)smem;
  _Float16* sWhh1 = sWih1 + N_WIH1;
  _Float16* sWih2 = sWhh1 + N_WHH1;
  _Float16* sWhh2 = sWih2 + N_WIH2;
  _Float16* sSt1  = sWhh2 + N_WHH2;
  _Float16* sSt2  = sSt1 + N_ST1;
  float*    sFin  = (float*)(sSt2 + N_ST2);
  float*    sXb   = sFin + N_FIN;

  const int tid = threadIdx.x;

  // One-time cooperative weight load: f32 global -> f16 LDS (padded strides).
  for (int i = tid; i < 256 * 64; i += blockDim.x)
    sWih1[(i / 64) * S_W1 + (i % 64)] = (_Float16)W_ih1[i];
  for (int i = tid; i < 256 * 64; i += blockDim.x)
    sWhh1[(i / 64) * S_W1 + (i % 64)] = (_Float16)W_hh1[i];
  for (int i = tid; i < 128 * 64; i += blockDim.x)
    sWih2[(i / 64) * S_W1 + (i % 64)] = (_Float16)W_ih2[i];
  for (int i = tid; i < 128 * 32; i += blockDim.x)
    sWhh2[(i / 32) * S_W2s + (i % 32)] = (_Float16)W_hh2[i];
  __syncthreads();

  const int wave = tid >> 5;
  const int lane = tid & 31;
  const int m    = lane & 15;   // batch row in tile / N-col within tile
  const int half = lane >> 4;

  const int batchBase = blockIdx.x * (WAVES * 16) + wave * 16;

  _Float16* st1 = sSt1 + wave * 16 * S_ST1;
  _Float16* st2 = sSt2 + wave * 16 * S_ST2;
  float*    fin = sFin + wave * 16 * 32;
  float*    xb0 = sXb + wave * 2 * 16 * S_XB;
  float*    xb1 = xb0 + 16 * S_XB;

  // Async x-tile addressing: lane -> (row r = lane&15, 16B-chunk parity h).
  // Each of the 8 issues moves 32 lanes * 16B = 512B; offsets step 32B so a
  // lane covers chunks {h, h+2, ...} of its row.
  const unsigned xgoff = (unsigned)(m * XROW_BYTES + half * 16);
  const unsigned xlds0 = (unsigned)(uintptr_t)(xb0 + m * S_XB) + half * 16;
  const unsigned xlds1 = (unsigned)(uintptr_t)(xb1 + m * S_XB) + half * 16;
  const char* xgbase = (const char*)x + (size_t)batchBase * XROW_BYTES;

  // Per-lane gate biases (column = tile*16 + m in the gate dimension).
  float bias1[16];
#pragma unroll
  for (int tt = 0; tt < 16; ++tt) {
    int n = tt * 16 + m;
    bias1[tt] = b_ih1[n] + b_hh1[n];
  }
  float bias2[8];
#pragma unroll
  for (int tt = 0; tt < 8; ++tt) {
    int n = tt * 16 + m;
    bias2[tt] = b_ih2[n] + b_hh2[n];
  }

  // Recurrent state: h kept as f16 A-fragments, c kept f32 in C/D layout.
  v16h h1A0 = {}, h1A1 = {}, h2A = {};
  v8f c1[4] = {};
  v8f c2[2] = {};

  // Prefetch x tile for t = 0 into buffer 0.
  async_x_tile(xgbase, xlds0, xgoff);

#pragma unroll 1
  for (int t = 0; t < Tlen; ++t) {
    // Wait for the current tile (issued last iteration / prologue).
    wait_async0();
    // Prefetch t+1 into the other buffer while we compute on this one.
    if (t + 1 < Tlen) {
      async_x_tile(xgbase + (size_t)(t + 1) * XT_BYTES,
                   ((t + 1) & 1) ? xlds1 : xlds0, xgoff);
    }
    const float* xb = (t & 1) ? xb1 : xb0;
    v16h xA0 = frag_x_from_lds(xb, m, 0, half);
    v16h xA1 = frag_x_from_lds(xb, m, 32, half);

    // ---- Layer 1: gates (256 wide) in 4 hidden chunks of 16 cols ----
#pragma unroll
    for (int ch = 0; ch < 4; ++ch) {
      v8f acc[4];
#pragma unroll
      for (int g = 0; g < 4; ++g) {
        const int tile = ch + 4 * g;           // i,f,g,o tiles share hidden cols
        const float bv = bias1[tile];
        v8f a = {bv, bv, bv, bv, bv, bv, bv, bv};
        a = wmma_f16(xA0,  frag_from_lds(sWih1, S_W1, tile * 16 + m, 0,  half), a);
        a = wmma_f16(xA1,  frag_from_lds(sWih1, S_W1, tile * 16 + m, 32, half), a);
        a = wmma_f16(h1A0, frag_from_lds(sWhh1, S_W1, tile * 16 + m, 0,  half), a);
        a = wmma_f16(h1A1, frag_from_lds(sWhh1, S_W1, tile * 16 + m, 32, half), a);
        acc[g] = a;
      }
#pragma unroll
      for (int r = 0; r < 8; ++r) {
        float iv = sigf(acc[0][r]);
        float fv = sigf(acc[1][r]);
        float gv = tanh_fast(acc[2][r]);
        float ov = sigf(acc[3][r]);
        float cn = fv * c1[ch][r] + iv * gv;
        c1[ch][r] = cn;
        float hv = ov * tanh_fast(cn);
        st1[(r + 8 * half) * S_ST1 + ch * 16 + m] = (_Float16)hv;  // D-layout -> LDS
      }
    }
    // Transpose h1_t back into A-operand layout via LDS (same-wave DS in-order).
    h1A0 = frag_from_lds(st1, S_ST1, m, 0,  half);
    h1A1 = frag_from_lds(st1, S_ST1, m, 32, half);

    // ---- Layer 2: gates (128 wide) in 2 hidden chunks ----
#pragma unroll
    for (int ch = 0; ch < 2; ++ch) {
      v8f acc[4];
#pragma unroll
      for (int g = 0; g < 4; ++g) {
        const int tile = ch + 2 * g;
        const float bv = bias2[tile];
        v8f a = {bv, bv, bv, bv, bv, bv, bv, bv};
        a = wmma_f16(h1A0, frag_from_lds(sWih2, S_W1,  tile * 16 + m, 0,  half), a);
        a = wmma_f16(h1A1, frag_from_lds(sWih2, S_W1,  tile * 16 + m, 32, half), a);
        a = wmma_f16(h2A,  frag_from_lds(sWhh2, S_W2s, tile * 16 + m, 0,  half), a);
        acc[g] = a;
      }
#pragma unroll
      for (int r = 0; r < 8; ++r) {
        float iv = sigf(acc[0][r]);
        float fv = sigf(acc[1][r]);
        float gv = tanh_fast(acc[2][r]);
        float ov = sigf(acc[3][r]);
        float cn = fv * c2[ch][r] + iv * gv;
        c2[ch][r] = cn;
        float hv = ov * tanh_fast(cn);
        st2[(r + 8 * half) * S_ST2 + ch * 16 + m] = (_Float16)hv;
        if (t == Tlen - 1)
          fin[(r + 8 * half) * 32 + ch * 16 + m] = hv;   // f32 copy for the MLP
      }
    }
    h2A = frag_from_lds(st2, S_ST2, m, 0, half);
  }

  // ---- Final MLP on h2[T-1]: one batch row per lane (lanes 0..15) ----
  if (lane < 16) {
    const float* hrow = fin + lane * 32;
    float a1[32];
#pragma unroll 4
    for (int j = 0; j < 32; ++j) {
      float s = b_fc1[j];
      for (int k = 0; k < 32; ++k) s += W_fc1[j * 32 + k] * hrow[k];
      a1[j] = fmaxf(s, 0.0f);
    }
    float a2[16];
#pragma unroll 4
    for (int j = 0; j < 16; ++j) {
      float s = b_fc2[j];
      for (int k = 0; k < 32; ++k) s += W_fc2[j * 32 + k] * a1[k];
      a2[j] = fmaxf(s, 0.0f);
    }
    float s = b_out[0];
#pragma unroll
    for (int k = 0; k < 16; ++k) s += W_out[k] * a2[k];
    out[batchBase + lane] = s;
  }
}

extern "C" void kernel_launch(void* const* d_in, const int* in_sizes, int n_in,
                              void* d_out, int out_size, void* d_ws, size_t ws_size,
                              hipStream_t stream) {
  (void)in_sizes; (void)n_in; (void)out_size; (void)d_ws; (void)ws_size;
  const float* x     = (const float*)d_in[0];
  const float* W_ih1 = (const float*)d_in[1];
  const float* W_hh1 = (const float*)d_in[2];
  const float* b_ih1 = (const float*)d_in[3];
  const float* b_hh1 = (const float*)d_in[4];
  const float* W_ih2 = (const float*)d_in[5];
  const float* W_hh2 = (const float*)d_in[6];
  const float* b_ih2 = (const float*)d_in[7];
  const float* b_hh2 = (const float*)d_in[8];
  const float* W_fc1 = (const float*)d_in[9];
  const float* b_fc1 = (const float*)d_in[10];
  const float* W_fc2 = (const float*)d_in[11];
  const float* b_fc2 = (const float*)d_in[12];
  const float* W_out = (const float*)d_in[13];
  const float* b_out = (const float*)d_in[14];
  float* out = (float*)d_out;

  // Unconditional (harness forbids static guards); deterministic + cheap.
  hipFuncSetAttribute((const void*)lstm2_fused_wmma,
                      hipFuncAttributeMaxDynamicSharedMemorySize,
                      (int)SMEM_BYTES);

  const int blocks = Bsz / (WAVES * 16);  // 64 blocks * 4 waves * 16 rows = 4096
  lstm2_fused_wmma<<<blocks, WAVES * 32, SMEM_BYTES, stream>>>(
      x, W_ih1, W_hh1, b_ih1, b_hh1, W_ih2, W_hh2, b_ih2, b_hh2,
      W_fc1, b_fc1, W_fc2, b_fc2, W_out, b_out, out);
}
